// FlowLayer_83494164234473
// MI455X (gfx1250) — matrix operand order repaired and checked by reading, not compiled, not run on hardware
//
#include <hip/hip_runtime.h>
#include <hip/hip_bf16.h>

typedef __attribute__((ext_vector_type(16))) _Float16     v16h;
typedef __attribute__((ext_vector_type(8)))  float        v8f;
typedef __attribute__((ext_vector_type(4)))  unsigned int v4u;
typedef __attribute__((ext_vector_type(8)))  int          v8i;
typedef __attribute__((ext_vector_type(4)))  int          v4i;

#define H_  128
#define W_  128
#define HW_ (128 * 128)

#if defined(__HIP_DEVICE_COMPILE__) && __has_builtin(__builtin_amdgcn_tensor_load_to_lds)
#define HAVE_TDM 1
#else
#define HAVE_TDM 0
#endif

__device__ __forceinline__ float sigmoidf(float v) {
    return 1.0f / (1.0f + __expf(-v));
}

// ---------------------------------------------------------------------------
// WMMA 1x1-conv GEMM:  Out[b][o][p] = act( sum_k Wrow(o)[k] * In[b][k][p] )
//   - Wrow(o) = W0 + o*K for o<64, else W1 + (o-64)*K   (one launch computes
//     m1 (ch 0..63) and m2 (ch 64..127) into a single 128-ch tensor)
//   - block = 256 threads (8 waves), 32-pixel strip per block
//   - activation tile staged via the Tensor Data Mover (tensor_load_to_lds,
//     2D descriptor, TENSORcnt), then converted once to an f16 [p][k] tile
//   - waves own 16x16 output tiles, K accumulated via v_wmma_f32_16x16x32_f16
// ---------------------------------------------------------------------------
template<int KDIM, int COUT, bool SIG>
__global__ __launch_bounds__(256) void gemm1x1_wmma(
    const float* __restrict__ W0, const float* __restrict__ W1,
    const float* __restrict__ In, float* __restrict__ Out)
{
#if HAVE_TDM
    __shared__ __align__(16) float    sXf[KDIM * 32];  // [k][p], f32 (TDM dest)
#endif
    __shared__ __align__(32) _Float16 sX[32 * KDIM];   // [p][k], f16 (WMMA B)

    const int b  = blockIdx.y;
    const int p0 = blockIdx.x * 32;
    const float* __restrict__ inb = In + (size_t)b * KDIM * HW_;

#if HAVE_TDM
    // --- TDM: DMA the 2D tile X[k=0..KDIM-1][p0..p0+31] (row stride HW_)
    //     from global into LDS as contiguous [k][p] f32. Wave 0 issues it.
    if ((threadIdx.x >> 5) == 0) {
        const unsigned long long ga =
            (unsigned long long)(size_t)(const void*)(inb + p0);
        const unsigned int lds_base = (unsigned int)(size_t)(void*)&sXf[0];

        v4u g0;
        g0[0] = 1u;                                       // count=1, user mode
        g0[1] = lds_base;                                 // lds_addr (bytes)
        g0[2] = (unsigned int)(ga & 0xFFFFFFFFull);       // global_addr[31:0]
        g0[3] = (unsigned int)((ga >> 32) & 0x01FFFFFFull)
              | (2u << 30);                               // type=2 ("image")

        v8i g1;
        g1[0] = (int)(2u << 16);                          // data_size = 4 bytes
        g1[1] = (int)((unsigned)HW_ << 16);               // tensor_dim0[15:0]
        g1[2] = (int)(((unsigned)HW_ >> 16)               // tensor_dim0[31:16]
              | ((unsigned)KDIM << 16));                  // tensor_dim1[15:0]
        g1[3] = (int)(32u << 16);                         // tile_dim0 = 32
        g1[4] = (int)KDIM;                                // tile_dim1 = KDIM
        g1[5] = (int)HW_;                                 // tensor_dim0_stride
        g1[6] = 0;
        g1[7] = 0;

        v4i gz = {0, 0, 0, 0};
#if __clang_major__ >= 23
        v8i gz8 = {0, 0, 0, 0, 0, 0, 0, 0};
        __builtin_amdgcn_tensor_load_to_lds(g0, g1, gz, gz, gz8, 0);
#else
        __builtin_amdgcn_tensor_load_to_lds(g0, g1, gz, gz, 0);
#endif
        __builtin_amdgcn_s_wait_tensorcnt((short)0);
    }
    __syncthreads();

    // convert [k][p] f32 -> [p][k] f16 for the WMMA B-fragment layout
    for (int idx = threadIdx.x; idx < 32 * KDIM; idx += 256) {
        const int p = idx & 31;
        const int k = idx >> 5;
        sX[p * KDIM + k] = (_Float16)sXf[k * 32 + p];
    }
#else
    // fallback: direct global staging (coalesced over pixels), f32 -> f16
    for (int idx = threadIdx.x; idx < 32 * KDIM; idx += 256) {
        const int p = idx & 31;
        const int k = idx >> 5;
        sX[p * KDIM + k] = (_Float16)inb[k * HW_ + p0 + p];
    }
#endif
    __syncthreads();

    const int wave = threadIdx.x >> 5;
    const int lane = threadIdx.x & 31;
    const int hl   = lane & 15;      // M (A) / N (B) index
    const int hi   = lane >> 4;      // lane-half select
    const int kb   = hi * 16;        // A-matrix K base per lane-half

    const int tiles = (COUT / 16) * 2;          // Mtiles x 2 pixel-subtiles
    for (int t = wave; t < tiles; t += 8) {
        const int m  = t >> 1;
        const int n  = t & 1;
        const int oc = m * 16 + hl;
        const float* __restrict__ wr =
            (oc < 64) ? (W0 + oc * KDIM) : (W1 + (oc - 64) * KDIM);

        v8f c = {0.f, 0.f, 0.f, 0.f, 0.f, 0.f, 0.f, 0.f};
        #pragma unroll
        for (int k0 = 0; k0 < KDIM; k0 += 32) {
            __builtin_prefetch(wr + k0 + 32, 0, 0);   // global_prefetch_b8
            // A: 16x32 f16, contiguous K order: halves 0..15 = K(k0+kb ..+15),
            //    kb = 16*hi (matches B)
            const float4 f0 = *(const float4*)(wr + k0 + kb);
            const float4 f1 = *(const float4*)(wr + k0 + kb + 4);
            const float4 f2 = *(const float4*)(wr + k0 + kb + 8);
            const float4 f3 = *(const float4*)(wr + k0 + kb + 12);
            v16h a;
            a[0]  = (_Float16)f0.x; a[1]  = (_Float16)f0.y;
            a[2]  = (_Float16)f0.z; a[3]  = (_Float16)f0.w;
            a[4]  = (_Float16)f1.x; a[5]  = (_Float16)f1.y;
            a[6]  = (_Float16)f1.z; a[7]  = (_Float16)f1.w;
            a[8]  = (_Float16)f2.x; a[9]  = (_Float16)f2.y;
            a[10] = (_Float16)f2.z; a[11] = (_Float16)f2.w;
            a[12] = (_Float16)f3.x; a[13] = (_Float16)f3.y;
            a[14] = (_Float16)f3.z; a[15] = (_Float16)f3.w;

            // B: 32x16, lane n holds 16 contiguous K halves (hi*16 split)
            const v16h bm =
                *(const v16h*)&sX[(n * 16 + hl) * KDIM + k0 + hi * 16];

            c = __builtin_amdgcn_wmma_f32_16x16x32_f16(
                    false, a, false, bm, (short)0, c, false, false);
        }

        // D layout: VGPR r -> M = r + 8*hi, N = hl
        float* __restrict__ orow =
            Out + ((size_t)b * COUT + (m * 16 + hi * 8)) * HW_ + p0 + n * 16 + hl;
        #pragma unroll
        for (int r = 0; r < 8; ++r) {
            float v = c[r];
            if (SIG) v = sigmoidf(v);
            orow[(size_t)r * HW_] = v;
        }
    }
}

// ---------------------------------------------------------------------------
// One advection step: Out[b][c][p] = In[b][c][p] + bilinear(In[b][c], +/-g)
// channels 0..63 use +grid, channels 64..127 use -grid (m1 / m2 fused).
// One thread per (b, pixel); gather offsets/weights computed once, reused
// over all 128 channels.
// ---------------------------------------------------------------------------
__device__ __forceinline__ void bilin_setup(float gx, float gy,
                                            int* off, float* wt)
{
    const float xs = (gx + 1.0f) * (W_ * 0.5f) - 0.5f;
    const float ys = (gy + 1.0f) * (H_ * 0.5f) - 0.5f;
    const float xf = floorf(xs), yf = floorf(ys);
    const float tx = xs - xf,  ty = ys - yf;
    const int x0 = (int)xf, y0 = (int)yf;
    const float wg[4] = { (1.f-ty)*(1.f-tx), (1.f-ty)*tx, ty*(1.f-tx), ty*tx };
    const int xi[4] = { x0, x0 + 1, x0,     x0 + 1 };
    const int yi[4] = { y0, y0,     y0 + 1, y0 + 1 };
    #pragma unroll
    for (int i = 0; i < 4; ++i) {
        const bool v = (xi[i] >= 0) & (xi[i] < W_) & (yi[i] >= 0) & (yi[i] < H_);
        const int xc = xi[i] < 0 ? 0 : (xi[i] > W_ - 1 ? W_ - 1 : xi[i]);
        const int yc = yi[i] < 0 ? 0 : (yi[i] > H_ - 1 ? H_ - 1 : yi[i]);
        off[i] = yc * W_ + xc;
        wt[i]  = v ? wg[i] : 0.0f;
    }
}

__global__ __launch_bounds__(256) void flow_step(
    const float* __restrict__ In, float* __restrict__ Out,
    const float* __restrict__ grads, float inv_iters)
{
    const int gid = blockIdx.x * 256 + threadIdx.x;
    const int b = gid >> 14;           // HW_ = 16384
    const int p = gid & (HW_ - 1);

    const float gx = grads[((size_t)b * 2 + 0) * HW_ + p] * inv_iters;
    const float gy = grads[((size_t)b * 2 + 1) * HW_ + p] * inv_iters;

    int   offP[4], offN[4];
    float wP[4],   wN[4];
    bilin_setup( gx,  gy, offP, wP);
    bilin_setup(-gx, -gy, offN, wN);

    const float* __restrict__ ib = In  + (size_t)b * 128 * HW_;
    float*       __restrict__ ob = Out + (size_t)b * 128 * HW_;

    #pragma unroll 4
    for (int c = 0; c < 64; ++c) {
        const float* ch = ib + (size_t)c * HW_;
        const float s = ch[offP[0]] * wP[0] + ch[offP[1]] * wP[1] +
                        ch[offP[2]] * wP[2] + ch[offP[3]] * wP[3];
        ob[(size_t)c * HW_ + p] = ch[p] + s;
    }
    #pragma unroll 4
    for (int c = 64; c < 128; ++c) {
        const float* ch = ib + (size_t)c * HW_;
        const float s = ch[offN[0]] * wN[0] + ch[offN[1]] * wN[1] +
                        ch[offN[2]] * wN[2] + ch[offN[3]] * wN[3];
        ob[(size_t)c * HW_ + p] = ch[p] + s;
    }
}

// ---------------------------------------------------------------------------
// Depthwise 3x3, zero padding; optional sigmoid fused on the input reads
// (merge stage applies sigmoid to the flow state before conv).
// One thread per (b, c, pixel).
// ---------------------------------------------------------------------------
template<bool SIG_IN>
__global__ __launch_bounds__(256) void dw3x3(
    const float* __restrict__ In, const float* __restrict__ Wd,
    float* __restrict__ Out, int C)
{
    const int gid = blockIdx.x * 256 + threadIdx.x;
    const int p  = gid & (HW_ - 1);
    const int bc = gid >> 14;
    const int c  = bc % C;
    const int py = p >> 7, px = p & (W_ - 1);

    const float* __restrict__ ib = In + (size_t)bc * HW_;
    const float* __restrict__ wk = Wd + c * 9;

    float acc = 0.0f;
    #pragma unroll
    for (int dy = -1; dy <= 1; ++dy) {
        #pragma unroll
        for (int dx = -1; dx <= 1; ++dx) {
            const int yy = py + dy, xx = px + dx;
            if (yy >= 0 && yy < H_ && xx >= 0 && xx < W_) {
                float v = ib[yy * W_ + xx];
                if (SIG_IN) v = sigmoidf(v);
                acc += wk[(dy + 1) * 3 + (dx + 1)] * v;
            }
        }
    }
    Out[(size_t)gid] = acc;
}

// ---------------------------------------------------------------------------
extern "C" void kernel_launch(void* const* d_in, const int* in_sizes, int n_in,
                              void* d_out, int out_size, void* d_ws, size_t ws_size,
                              hipStream_t stream)
{
    const float* x        = (const float*)d_in[0];
    const float* grads    = (const float*)d_in[1];
    const float* w1       = (const float*)d_in[2];
    const float* w2       = (const float*)d_in[3];
    const float* merge_dw = (const float*)d_in[4];
    const float* merge_pw = (const float*)d_in[5];
    const float* post_dw  = (const float*)d_in[6];
    const float* post_pw  = (const float*)d_in[7];
    float* out = (float*)d_out;

    const int B = in_sizes[1] / (2 * HW_);     // 4

    // workspace: two ping-pong (B,128,HW) f32 tensors = 64 MB total
    float* M0 = (float*)d_ws;
    float* M1 = M0 + (size_t)B * 128 * HW_;

    const dim3 gg(HW_ / 32, B);

    // 1) m1 = sigmoid(x @ w1^T) -> ch 0..63 ; m2 = sigmoid(x @ w2^T) -> ch 64..127
    gemm1x1_wmma<256, 128, true><<<gg, 256, 0, stream>>>(w1, w2, x, M0);

    // 2) 16 advection steps (M0 -> M1 -> M0 -> ... ends in M0)
    const int nthr = B * HW_;
    const float inv_iters = 1.0f / 16.0f;
    float* src = M0;
    float* dst = M1;
    for (int s = 0; s < 16; ++s) {
        flow_step<<<nthr / 256, 256, 0, stream>>>(src, dst, grads, inv_iters);
        float* t = src; src = dst; dst = t;
    }

    // 3) sigmoid + concat (free: already one 128-ch tensor) + merge depthwise
    dw3x3<true><<<(B * 128 * HW_) / 256, 256, 0, stream>>>(M0, merge_dw, M1, 128);

    // 4) merge pointwise 128 -> 64  (WMMA GEMM) -> M0
    gemm1x1_wmma<128, 64, false><<<gg, 256, 0, stream>>>(merge_pw, merge_pw, M1, M0);

    // 5) post depthwise 3x3 (64 ch) -> M1
    dw3x3<false><<<(B * 64 * HW_) / 256, 256, 0, stream>>>(M0, post_dw, M1, 64);

    // 6) post pointwise 64 -> 64 (WMMA GEMM) -> final output
    gemm1x1_wmma<64, 64, false><<<gg, 256, 0, stream>>>(post_pw, post_pw, M1, out);
}
